// RNNModel_62517543960781
// MI455X (gfx1250) — compile-verified, matrix operand order checked by reading, not run on hardware
//
#include <hip/hip_runtime.h>
#include <cstdint>
#include <cstddef>

// ---------------------------------------------------------------------------
// LSTM (B=64,T=512,I=256,H=512) + FC(C=40) for gfx1250.
// Phase 1: xg[t][b][g] = X @ W_ih^T + (b_ih+b_hh)  via bf16 WMMA, f32 accum.
// Phase 2: persistent 32-WG scan; W_hh slice LDS-resident, software-pipelined
//          bf16 WMMA recurrent GEMM, one device barrier per timestep.
// Phase 3: tiny fp32 FC head.
// ---------------------------------------------------------------------------

typedef __attribute__((ext_vector_type(16))) __bf16 v16bf;
typedef __attribute__((ext_vector_type(8)))  float  v8f;

union Frag16 { uint4 u[2]; v16bf v; };

#define B_   64
#define T_   512
#define I_   256
#define H_   512
#define G_   2048   // 4*H
#define C_   40
#define NWG  32     // workgroups in scan kernel
#define HC   16     // hidden units per scan workgroup
#define LROW 64     // 4*HC gate rows per scan workgroup

static __device__ __forceinline__ unsigned short f32_to_bf16(float f) {
  unsigned int u = __float_as_uint(f);
  u = (u + 0x7FFFu + ((u >> 16) & 1u)) >> 16;
  return (unsigned short)u;
}

// ---------------- fp32 -> bf16 conversion (x4 vectorized) ------------------
__global__ __launch_bounds__(256) void cvt_f32_bf16x4(
    const float4* __restrict__ src, ushort4* __restrict__ dst, int n4) {
  int i = blockIdx.x * 256 + threadIdx.x;
  if (i < n4) {
    float4 f = src[i];
    ushort4 o;
    o.x = f32_to_bf16(f.x); o.y = f32_to_bf16(f.y);
    o.z = f32_to_bf16(f.z); o.w = f32_to_bf16(f.w);
    dst[i] = o;
  }
}

// ---------------- state init (h=0 double buffer, barrier=0) ----------------
__global__ __launch_bounds__(256) void init_state(
    unsigned short* __restrict__ hbf, unsigned int* __restrict__ bar, int n) {
  int i = blockIdx.x * 256 + threadIdx.x;
  if (i < n) hbf[i] = 0;
  if (i == 0) *bar = 0u;
}

// ---------------- xg[t][b][g] = X @ W_ih^T + (b_ih + b_hh) -----------------
// Grid: 256 blocks (128 rows of BT each), block 256 threads = 8 waves.
// Wave w owns M-tile w; A fragments for all 8 k-steps cached in registers.
// W_ih staged in LDS in 64-column chunks; per k-step all 4 B fragments are
// loaded before the 4 WMMAs so LDS waits overlap the matrix pipe.
__global__ __launch_bounds__(256) void xg_gemm(
    const unsigned short* __restrict__ xbf,    // [B*T, 256] bf16
    const unsigned short* __restrict__ wihbf,  // [2048, 256] bf16
    const float* __restrict__ b_ih,
    const float* __restrict__ b_hh,
    float* __restrict__ xg) {                  // [T, B, 2048] f32
  __shared__ unsigned short wlds[64 * 264];    // 64 rows, padded stride

  const int tid  = threadIdx.x;
  const int wv   = tid >> 5;
  const int lane = tid & 31;
  const int half = lane >> 4;
  const int mn   = lane & 15;

  // Preload A: this wave's 16 rows x full K=256 (8 k-steps x 2 chunks).
  const int rowA = blockIdx.x * 128 + wv * 16 + mn;
  Frag16 afrag[8];
  for (int ks = 0; ks < 8; ++ks) {
    const int k0 = ks * 32;
    afrag[ks].u[0] = *(const uint4*)(xbf + (size_t)rowA * I_ + k0 + half * 8);
    afrag[ks].u[1] = *(const uint4*)(xbf + (size_t)rowA * I_ + k0 + 16 + half * 8);
  }

  for (int nc = 0; nc < 32; ++nc) {            // 32 chunks of 64 gate cols
    const int nbase = nc * 64;
    __syncthreads();
    // Stage W_ih[nbase:nbase+64, 0:256] -> LDS (2048 uint4)
    for (int j = 0; j < 8; ++j) {
      const int u  = j * 256 + tid;
      const int r  = u >> 5;
      const int cq = u & 31;
      uint4 v = *(const uint4*)(wihbf + (size_t)(nbase + r) * I_ + cq * 8);
      *(uint4*)(wlds + r * 264 + cq * 8) = v;
    }
    __syncthreads();

    v8f acc[4];
    for (int nt = 0; nt < 4; ++nt)
      for (int v = 0; v < 8; ++v) acc[nt][v] = 0.f;

    #pragma unroll
    for (int ks = 0; ks < 8; ++ks) {
      const int k0 = ks * 32;
      Frag16 bf[4];
      #pragma unroll
      for (int nt = 0; nt < 4; ++nt) {
        const int lrow = nt * 16 + mn;
        bf[nt].u[0] = *(const uint4*)(wlds + lrow * 264 + k0 + half * 8);
        bf[nt].u[1] = *(const uint4*)(wlds + lrow * 264 + k0 + 16 + half * 8);
      }
      #pragma unroll
      for (int nt = 0; nt < 4; ++nt)
        acc[nt] = __builtin_amdgcn_wmma_f32_16x16x32_bf16(
            false, afrag[ks].v, false, bf[nt].v, (short)0, acc[nt], false, false);
    }

    for (int nt = 0; nt < 4; ++nt) {
      const int gcol = nbase + nt * 16 + mn;
      const float bias = b_ih[gcol] + b_hh[gcol];
      for (int v = 0; v < 8; ++v) {
        const int row = blockIdx.x * 128 + wv * 16 + v + 8 * half;  // bt index
        const int t = row & (T_ - 1);
        const int b = row >> 9;
        xg[((size_t)t * B_ + b) * G_ + gcol] = acc[nt][v] + bias;
      }
    }
  }
}

// ---------------- persistent LSTM scan -------------------------------------
// Grid: exactly NWG=32 blocks (co-resident), block 256 threads = 8 waves.
// WG k owns hidden units [k*HC,(k+1)*HC): its 4*HC W_hh rows stay in LDS for
// all 512 steps. h double-buffered bf16 in global (L2-hot); the per-step
// 64x64x512 GEMM is software-pipelined: fragments for k+32 are fetched while
// the two WMMAs for k execute; xg gate offsets are loaded at step start and
// only consumed after the k-loop (latency hidden under WMMA).
__global__ __launch_bounds__(256) void lstm_scan(
    const float* __restrict__ xg,              // [T, B, 2048] f32
    unsigned short* __restrict__ hbf,          // [2][64*512] bf16
    float* __restrict__ hf32,                  // [64*512] f32 (final h)
    const unsigned short* __restrict__ whhbf,  // [2048, 512] bf16
    unsigned int* __restrict__ bar) {
  __shared__ unsigned short whh[LROW * 520];   // 66,560 B (padded stride)
  __shared__ float gates[B_ * 68];             // 17,408 B

  const int tid = threadIdx.x;
  const int wg  = blockIdx.x;

  // Load this WG's W_hh slice: local row r -> global row (r/HC)*H + wg*HC + r%HC
  for (int j = 0; j < 16; ++j) {
    const int u    = j * 256 + tid;
    const int r    = u >> 6;
    const int cq   = u & 63;
    const int grow = (r >> 4) * H_ + wg * HC + (r & 15);
    uint4 v = *(const uint4*)(whhbf + (size_t)grow * H_ + cq * 8);
    *(uint4*)(whh + r * 520 + cq * 8) = v;
  }
  float creg[4] = {0.f, 0.f, 0.f, 0.f};
  __syncthreads();

  const int wv = tid >> 5, lane = tid & 31;
  const int half = lane >> 4, mn = lane & 15;
  const int mTile = wv >> 1;
  const int nt0   = (wv & 1) * 2;
  const int rowA  = mTile * 16 + mn;
  const int lrow0 = nt0 * 16 + mn;             // B rows in LDS (gate nt0, nt0+1)
  const int lrow1 = lrow0 + 16;
  const int g0    = nt0 * H_ + wg * HC + mn;   // xg columns for this lane

  for (int t = 0; t < T_; ++t) {
    const unsigned short* hread  = hbf + (t & 1) * (B_ * H_);
    unsigned short*       hwrite = hbf + ((t + 1) & 1) * (B_ * H_);
    const float* xgt = xg + (size_t)t * (B_ * G_);

    // Issue the 16 xg loads now; they complete under the k-loop.
    float xv[2][8];
    #pragma unroll
    for (int q = 0; q < 2; ++q)
      #pragma unroll
      for (int v = 0; v < 8; ++v)
        xv[q][v] = xgt[(size_t)(mTile * 16 + v + 8 * half) * G_ + g0 + q * H_];

    v8f acc[2];
    #pragma unroll
    for (int q = 0; q < 2; ++q)
      for (int v = 0; v < 8; ++v) acc[q][v] = 0.f;

    // gates = h @ W_hh_slice^T, software-pipelined over K=512.
    Frag16 A, Ba, Bb;
    A.u[0]  = *(const uint4*)(hread + rowA * H_ + half * 8);
    A.u[1]  = *(const uint4*)(hread + rowA * H_ + 16 + half * 8);
    Ba.u[0] = *(const uint4*)(whh + lrow0 * 520 + half * 8);
    Ba.u[1] = *(const uint4*)(whh + lrow0 * 520 + 16 + half * 8);
    Bb.u[0] = *(const uint4*)(whh + lrow1 * 520 + half * 8);
    Bb.u[1] = *(const uint4*)(whh + lrow1 * 520 + 16 + half * 8);
    #pragma unroll
    for (int ks = 0; ks < 16; ++ks) {
      Frag16 An = A, Ban = Ba, Bbn = Bb;
      if (ks < 15) {
        const int kn = (ks + 1) * 32;
        An.u[0]  = *(const uint4*)(hread + rowA * H_ + kn + half * 8);
        An.u[1]  = *(const uint4*)(hread + rowA * H_ + kn + 16 + half * 8);
        Ban.u[0] = *(const uint4*)(whh + lrow0 * 520 + kn + half * 8);
        Ban.u[1] = *(const uint4*)(whh + lrow0 * 520 + kn + 16 + half * 8);
        Bbn.u[0] = *(const uint4*)(whh + lrow1 * 520 + kn + half * 8);
        Bbn.u[1] = *(const uint4*)(whh + lrow1 * 520 + kn + 16 + half * 8);
      }
      acc[0] = __builtin_amdgcn_wmma_f32_16x16x32_bf16(
          false, A.v, false, Ba.v, (short)0, acc[0], false, false);
      acc[1] = __builtin_amdgcn_wmma_f32_16x16x32_bf16(
          false, A.v, false, Bb.v, (short)0, acc[1], false, false);
      A = An; Ba = Ban; Bb = Bbn;
    }

    // Scatter gate pre-activations (+xg, hidden under the k-loop) to LDS.
    #pragma unroll
    for (int q = 0; q < 2; ++q) {
      const int col = (nt0 + q) * 16 + mn;
      #pragma unroll
      for (int v = 0; v < 8; ++v) {
        const int b = mTile * 16 + v + 8 * half;
        gates[b * 68 + col] = acc[q][v] + xv[q][v];
      }
    }
    __syncthreads();

    // c,h update: 64 batches x HC hidden = 1024 pairs, 4 per thread.
    for (int j = 0; j < 4; ++j) {
      const int p  = j * 256 + tid;
      const int b  = p >> 4;
      const int hl = p & 15;
      float ig = gates[b * 68 + hl];
      float fg = gates[b * 68 + 16 + hl];
      float gg = gates[b * 68 + 32 + hl];
      float og = gates[b * 68 + 48 + hl];
      ig = 1.f / (1.f + __expf(-ig));
      fg = 1.f / (1.f + __expf(-fg));
      og = 1.f / (1.f + __expf(-og));
      gg = tanhf(gg);
      const float c = fg * creg[j] + ig * gg;
      creg[j] = c;
      const float h = og * tanhf(c);
      hwrite[b * H_ + wg * HC + hl] = f32_to_bf16(h);
      if (t == T_ - 1) hf32[b * H_ + wg * HC + hl] = h;
    }
    __threadfence();
    __syncthreads();

    // Device-scope grid barrier (monotonic counter; reset each launch).
    if (tid == 0) {
      __hip_atomic_fetch_add(bar, 1u, __ATOMIC_RELEASE, __HIP_MEMORY_SCOPE_AGENT);
      const unsigned int target = (unsigned int)(NWG * (t + 1));
      while (__hip_atomic_load(bar, __ATOMIC_ACQUIRE, __HIP_MEMORY_SCOPE_AGENT) < target) {
      }
    }
    __syncthreads();
  }
}

// ---------------- out = h_last @ W_fc^T + b_fc -----------------------------
__global__ __launch_bounds__(256) void fc_head(
    const float* __restrict__ h, const float* __restrict__ wfc,
    const float* __restrict__ bfc, float* __restrict__ out) {
  const int idx = blockIdx.x * 256 + threadIdx.x;
  if (idx >= B_ * C_) return;
  const int b = idx / C_, c = idx % C_;
  const float4* hp = (const float4*)(h + (size_t)b * H_);
  const float4* wp = (const float4*)(wfc + (size_t)c * H_);
  float s = bfc[c];
  for (int k = 0; k < H_ / 4; ++k) {
    float4 a = hp[k], w = wp[k];
    s += a.x * w.x + a.y * w.y + a.z * w.z + a.w * w.w;
  }
  out[idx] = s;
}

// ---------------- host launcher --------------------------------------------
extern "C" void kernel_launch(void* const* d_in, const int* in_sizes, int n_in,
                              void* d_out, int out_size, void* d_ws, size_t ws_size,
                              hipStream_t stream) {
  const float* x    = (const float*)d_in[0];   // [64,512,256]
  const float* Wih  = (const float*)d_in[1];   // [2048,256]
  const float* Whh  = (const float*)d_in[2];   // [2048,512]
  const float* bih  = (const float*)d_in[3];   // [2048]
  const float* bhh  = (const float*)d_in[4];   // [2048]
  const float* Wfc  = (const float*)d_in[5];   // [40,512]
  const float* bfc  = (const float*)d_in[6];   // [40]
  float* out = (float*)d_out;                  // [64,40]

  // Workspace layout (bytes), 256-aligned. Total ~288.7 MB.
  char* ws = (char*)d_ws;
  unsigned short* xbf   = (unsigned short*)(ws + 0);           // 16,777,216
  unsigned short* wihbf = (unsigned short*)(ws + 16777216);    //  1,048,576
  unsigned short* whhbf = (unsigned short*)(ws + 17825792);    //  2,097,152
  float*          xg    = (float*)         (ws + 19922944);    // 268,435,456
  unsigned short* hbf   = (unsigned short*)(ws + 288358400);   //    131,072 (x2 buf)
  float*          hf32  = (float*)         (ws + 288489472);   //    131,072
  unsigned int*   bar   = (unsigned int*)  (ws + 288620544);   //        256

  // 1) bf16 conversions (x4 vectorized; all sizes divisible by 4)
  cvt_f32_bf16x4<<<(B_ * T_ * I_ / 4 + 255) / 256, 256, 0, stream>>>(
      (const float4*)x, (ushort4*)xbf, B_ * T_ * I_ / 4);
  cvt_f32_bf16x4<<<(G_ * I_ / 4 + 255) / 256, 256, 0, stream>>>(
      (const float4*)Wih, (ushort4*)wihbf, G_ * I_ / 4);
  cvt_f32_bf16x4<<<(G_ * H_ / 4 + 255) / 256, 256, 0, stream>>>(
      (const float4*)Whh, (ushort4*)whhbf, G_ * H_ / 4);

  // 2) zero h double-buffer + barrier counter (re-done every launch)
  init_state<<<(2 * B_ * H_ + 255) / 256, 256, 0, stream>>>(hbf, bar, 2 * B_ * H_);

  // 3) input projection GEMM (WMMA bf16) -> xg[t][b][g]
  xg_gemm<<<(B_ * T_) / 128, 256, 0, stream>>>(xbf, wihbf, bih, bhh, xg);

  // 4) persistent recurrent scan (WMMA bf16, LDS-resident W_hh)
  lstm_scan<<<NWG, 256, 0, stream>>>(xg, hbf, hf32, whhbf, bar);

  // 5) FC head
  fc_head<<<(B_ * C_ + 255) / 256, 256, 0, stream>>>(hf32, Wfc, bfc, out);
}
